// GraphNet_5746666242334
// MI455X (gfx1250) — compile-verified, matrix operand order checked by reading, not run on hardware
//
#include <hip/hip_runtime.h>

// ---------------------------------------------------------------------------
// GraphNet (2-step message passing, L=128) for gfx1250 / MI455X.
// GEMMs via v_wmma_f32_16x16x32_bf16 (wave32 WMMA), gathers/MLP/scatter fused.
// ---------------------------------------------------------------------------

typedef __bf16 bf16_t;
typedef __attribute__((ext_vector_type(16))) __bf16 v16bf;
typedef __attribute__((ext_vector_type(8)))  __bf16 v8bf;
typedef __attribute__((ext_vector_type(8)))  float  v8f;

#define LAT 128   // latent width

// round-to-nearest-even f32 -> bf16 (bit-level, no reliance on fcvt support)
__device__ __forceinline__ bf16_t f2bf(float f) {
  union { float f; unsigned u; } x; x.f = f;
  unsigned r = (x.u + 0x7FFFu + ((x.u >> 16) & 1u)) >> 16;
  union { unsigned short s; bf16_t b; } y; y.s = (unsigned short)r;
  return y.b;
}

// Build one wave-lane's 16-element WMMA fragment (A or B^T) from a row-major
// bf16 buffer. ISA 16-bit A layout: lane<16 holds K {0..7,16..23} of row
// M=lane; lane>=16 holds K {8..15,24..31} of row M=lane-16. koff = 0 or 8.
// Two contiguous 16B loads per lane (b128-friendly).
__device__ __forceinline__ v16bf load_frag(const bf16_t* p, int koff) {
  union { v16bf v; v8bf h[2]; } u;
  u.h[0] = *(const v8bf*)(p + koff);
  u.h[1] = *(const v8bf*)(p + koff + 16);
  return u.v;
}

// Two-layer MLP on a 16-row tile: H = relu(X@W0+b0); out = relu(H@W1+b1).
// Xs: [16,KIN] bf16 LDS. Hs: [16,128] bf16 LDS. Wt0: [128,KIN] bf16 global
// (pre-transposed), Wt1: [128,128]. Each of 8 waves owns 16 output columns.
// Returns the 8 accumulator rows for this lane (D-layout: lane<16 -> N=lane,
// M=r; lane>=16 -> N=lane-16, M=8+r).
template <int KIN>
__device__ __forceinline__ void mlp2_tile(
    const bf16_t* __restrict__ Xs, bf16_t* __restrict__ Hs,
    const bf16_t* __restrict__ Wt0, const float* __restrict__ b0,
    const bf16_t* __restrict__ Wt1, const float* __restrict__ b1,
    float out[8], int wave, int lane)
{
  const int cb   = wave * 16;
  const int col  = lane & 15;
  const int half = lane >> 4;
  const int koff = half * 8;

  v8f acc = {};
  #pragma unroll 4
  for (int kc = 0; kc < KIN; kc += 32) {
    v16bf a = load_frag(&Xs[(lane & 15) * KIN + kc], koff);
    v16bf b = load_frag(&Wt0[(cb + col) * KIN + kc], koff);
    acc = __builtin_amdgcn_wmma_f32_16x16x32_bf16(false, a, false, b,
                                                  (short)0, acc, false, false);
  }
  const float bias0 = b0[cb + col];
  #pragma unroll
  for (int r = 0; r < 8; ++r) {
    float v = acc[r] + bias0;
    v = v > 0.f ? v : 0.f;
    Hs[(half * 8 + r) * LAT + cb + col] = f2bf(v);
  }
  __syncthreads();

  v8f acc2 = {};
  #pragma unroll
  for (int kc = 0; kc < LAT; kc += 32) {
    v16bf a = load_frag(&Hs[(lane & 15) * LAT + kc], koff);
    v16bf b = load_frag(&Wt1[(cb + col) * LAT + kc], koff);
    acc2 = __builtin_amdgcn_wmma_f32_16x16x32_bf16(false, a, false, b,
                                                   (short)0, acc2, false, false);
  }
  const float bias1 = b1[cb + col];
  #pragma unroll
  for (int r = 0; r < 8; ++r) {
    float v = acc2[r] + bias1;
    out[r] = v > 0.f ? v : 0.f;
  }
}

// ---------------- weight prep: fp32 [K,128] -> bf16 [128,K] ----------------
__global__ void gn_cast_transpose(const float* __restrict__ W,
                                  bf16_t* __restrict__ Wt, int K) {
  int idx = blockIdx.x * 256 + threadIdx.x;
  if (idx < K * LAT) {
    int k = idx / LAT, c = idx % LAT;
    Wt[c * K + k] = f2bf(W[idx]);
  }
}

__global__ void gn_zero(float* __restrict__ p, int n) {
  int i = blockIdx.x * 256 + threadIdx.x;
  if (i < n) p[i] = 0.f;
}

// ---------------- embeddings (single GEMM, no relu), K in {32,64} ----------
__global__ void __launch_bounds__(256) gn_embed(
    const float* __restrict__ X, const bf16_t* __restrict__ Wt,
    const float* __restrict__ b, float* __restrict__ out, int M, int K)
{
  __shared__ __align__(16) bf16_t Xs[16 * 64];
  const int base = blockIdx.x * 16;
  const int tid  = threadIdx.x;
  for (int idx = tid; idx < 16 * K; idx += 256) {
    int row = idx / K, c = idx % K;
    int m = base + row;
    Xs[row * K + c] = f2bf(m < M ? X[(size_t)m * K + c] : 0.f);
  }
  __syncthreads();
  const int wave = tid >> 5, lane = tid & 31;
  const int cb = wave * 16, col = lane & 15, half = lane >> 4, koff = half * 8;
  v8f acc = {};
  for (int kc = 0; kc < K; kc += 32) {
    v16bf a  = load_frag(&Xs[(lane & 15) * K + kc], koff);
    v16bf bb = load_frag(&Wt[(cb + col) * K + kc], koff);
    acc = __builtin_amdgcn_wmma_f32_16x16x32_bf16(false, a, false, bb,
                                                  (short)0, acc, false, false);
  }
  const float bias = b[cb + col];
  #pragma unroll
  for (int r = 0; r < 8; ++r) {
    int m = base + half * 8 + r;
    if (m < M) out[(size_t)m * LAT + cb + col] = acc[r] + bias;
  }
}

__global__ void gn_glob_embed(const float* __restrict__ g,
                              const float* __restrict__ W,
                              const float* __restrict__ b,
                              float* __restrict__ glob) {
  int c = threadIdx.x;
  float acc = b[c];
  for (int k = 0; k < 16; ++k) acc = fmaf(g[k], W[k * LAT + c], acc);
  glob[c] = acc;
}

// ---------------- fused edge update: gather -> MLP2 -> scatter-add ---------
__global__ void __launch_bounds__(256) gn_edge_mlp(
    const float* __restrict__ edges, const float* __restrict__ nodes,
    const float* __restrict__ glob,
    const int* __restrict__ senders, const int* __restrict__ receivers,
    const bf16_t* __restrict__ Wt0, const float* __restrict__ b0,
    const bf16_t* __restrict__ Wt1, const float* __restrict__ b1,
    float* __restrict__ e_new, float* __restrict__ sent,
    float* __restrict__ recv, float* __restrict__ edge_agg, int E)
{
  __shared__ __align__(16) bf16_t Xs[16 * 512];
  __shared__ __align__(16) bf16_t Hs[16 * LAT];
  const int base = blockIdx.x * 16;
  const int tid  = threadIdx.x;
  __builtin_prefetch(Wt0, 0, 1);   // global_prefetch_b8; weights are L2-hot

  // Stage concat(edges[e], nodes[snd], nodes[rcv], glob) as bf16 in LDS.
  for (int idx = tid; idx < 16 * 512; idx += 256) {
    const int row = idx >> 9, c = idx & 511;
    const int e = base + row;
    float v = 0.f;
    if (e < E) {
      if (c < 128)      v = edges[(size_t)e * LAT + c];
      else if (c < 256) v = nodes[(size_t)senders[e]   * LAT + (c - 128)];
      else if (c < 384) v = nodes[(size_t)receivers[e] * LAT + (c - 256)];
      else              v = glob[c - 384];
    }
    Xs[row * 512 + c] = f2bf(v);
  }
  __syncthreads();

  const int wave = tid >> 5, lane = tid & 31;
  float out[8];
  mlp2_tile<512>(Xs, Hs, Wt0, b0, Wt1, b1, out, wave, lane);

  const int cb = wave * 16, col = lane & 15, half = lane >> 4;
  float colsum = 0.f;
  #pragma unroll
  for (int r = 0; r < 8; ++r) {
    const int e = base + half * 8 + r;
    if (e < E) {
      const float v = out[r];
      e_new[(size_t)e * LAT + cb + col] = v;
      atomicAdd(&sent[(size_t)senders[e]   * LAT + cb + col], v);
      atomicAdd(&recv[(size_t)receivers[e] * LAT + cb + col], v);
      colsum += v;
    }
  }
  colsum += __shfl_xor(colsum, 16, 32);       // fold the two M-halves
  if (half == 0) atomicAdd(&edge_agg[cb + col], colsum);
}

// ---------------- fused node update ----------------------------------------
__global__ void __launch_bounds__(256) gn_node_mlp(
    const float* __restrict__ nodes, const float* __restrict__ sent,
    const float* __restrict__ recv, const float* __restrict__ glob,
    const bf16_t* __restrict__ Wt0, const float* __restrict__ b0,
    const bf16_t* __restrict__ Wt1, const float* __restrict__ b1,
    float* __restrict__ n_new, float* __restrict__ node_agg, int N)
{
  __shared__ __align__(16) bf16_t Xs[16 * 512];
  __shared__ __align__(16) bf16_t Hs[16 * LAT];
  const int base = blockIdx.x * 16;
  const int tid  = threadIdx.x;
  __builtin_prefetch(Wt0, 0, 1);

  for (int idx = tid; idx < 16 * 512; idx += 256) {
    const int row = idx >> 9, c = idx & 511;
    const int n = base + row;
    float v = 0.f;
    if (n < N) {
      if (c < 128)      v = nodes[(size_t)n * LAT + c];
      else if (c < 256) v = sent[(size_t)n * LAT + (c - 128)];
      else if (c < 384) v = recv[(size_t)n * LAT + (c - 256)];
      else              v = glob[c - 384];
    }
    Xs[row * 512 + c] = f2bf(v);
  }
  __syncthreads();

  const int wave = tid >> 5, lane = tid & 31;
  float out[8];
  mlp2_tile<512>(Xs, Hs, Wt0, b0, Wt1, b1, out, wave, lane);

  const int cb = wave * 16, col = lane & 15, half = lane >> 4;
  float colsum = 0.f;
  #pragma unroll
  for (int r = 0; r < 8; ++r) {
    const int n = base + half * 8 + r;
    if (n < N) {
      const float v = out[r];
      n_new[(size_t)n * LAT + cb + col] = v;
      colsum += v;
    }
  }
  colsum += __shfl_xor(colsum, 16, 32);
  if (half == 0) atomicAdd(&node_agg[cb + col], colsum);
}

// ---------------- residual + LayerNorm (one wave32 per row) ----------------
__global__ void __launch_bounds__(256) gn_ln_residual(
    float* __restrict__ state, const float* __restrict__ delta,
    const float* __restrict__ sc, const float* __restrict__ bi, int M)
{
  const int row  = blockIdx.x * 8 + (threadIdx.x >> 5);
  const int lane = threadIdx.x & 31;
  if (row >= M) return;
  float v[4]; float sum = 0.f;
  #pragma unroll
  for (int i = 0; i < 4; ++i) {
    const int c = lane * 4 + i;
    v[i] = state[(size_t)row * LAT + c] + delta[(size_t)row * LAT + c];
    sum += v[i];
  }
  #pragma unroll
  for (int off = 16; off > 0; off >>= 1) sum += __shfl_xor(sum, off, 32);
  const float mu = sum * (1.f / 128.f);
  float var = 0.f;
  #pragma unroll
  for (int i = 0; i < 4; ++i) { const float d = v[i] - mu; var += d * d; }
  #pragma unroll
  for (int off = 16; off > 0; off >>= 1) var += __shfl_xor(var, off, 32);
  const float inv = rsqrtf(var * (1.f / 128.f) + 1e-6f);
  #pragma unroll
  for (int i = 0; i < 4; ++i) {
    const int c = lane * 4 + i;
    state[(size_t)row * LAT + c] = (v[i] - mu) * inv * sc[c] + bi[c];
  }
}

// ---------------- global branch: MLP2 + residual + LN (M=1, scalar) --------
__global__ void __launch_bounds__(128) gn_glob_update(
    float* __restrict__ glob, const float* __restrict__ node_agg,
    const float* __restrict__ edge_agg,
    const float* __restrict__ W0, const float* __restrict__ b0,
    const float* __restrict__ W1, const float* __restrict__ b1,
    const float* __restrict__ ln_s, const float* __restrict__ ln_b)
{
  __shared__ float xin[384];
  __shared__ float h[128];
  __shared__ float red[128];
  const int c = threadIdx.x;
  const float gold = glob[c];
  xin[c] = node_agg[c]; xin[128 + c] = edge_agg[c]; xin[256 + c] = gold;
  __syncthreads();
  float acc = b0[c];
  for (int k = 0; k < 384; ++k) acc = fmaf(xin[k], W0[k * LAT + c], acc);
  h[c] = fmaxf(acc, 0.f);
  __syncthreads();
  float acc2 = b1[c];
  for (int k = 0; k < 128; ++k) acc2 = fmaf(h[k], W1[k * LAT + c], acc2);
  const float g = fmaxf(acc2, 0.f) + gold;          // residual
  red[c] = g;
  __syncthreads();
  for (int s = 64; s > 0; s >>= 1) { if (c < s) red[c] += red[c + s]; __syncthreads(); }
  const float mu = red[0] * (1.f / 128.f);
  __syncthreads();
  const float d = g - mu;
  red[c] = d * d;
  __syncthreads();
  for (int s = 64; s > 0; s >>= 1) { if (c < s) red[c] += red[c + s]; __syncthreads(); }
  const float var = red[0] * (1.f / 128.f);
  glob[c] = d * rsqrtf(var + 1e-6f) * ln_s[c] + ln_b[c];
}

__global__ void gn_decode(const float* __restrict__ glob,
                          const float* __restrict__ W,
                          const float* __restrict__ b,
                          float* __restrict__ out) {
  int c = threadIdx.x;
  float acc = b[c];
  for (int k = 0; k < 128; ++k) acc = fmaf(glob[k], W[k * LAT + c], acc);
  out[c] = acc;
}

// ---------------------------------------------------------------------------
extern "C" void kernel_launch(void* const* d_in, const int* in_sizes, int n_in,
                              void* d_out, int out_size, void* d_ws, size_t ws_size,
                              hipStream_t stream)
{
  const float* node_feats = (const float*)d_in[0];
  const float* edge_feats = (const float*)d_in[1];
  const float* globals_   = (const float*)d_in[2];
  const int*   senders    = (const int*)d_in[3];
  const int*   receivers  = (const int*)d_in[4];
  const float* emb_node_W = (const float*)d_in[5];
  const float* emb_node_b = (const float*)d_in[6];
  const float* emb_edge_W = (const float*)d_in[7];
  const float* emb_edge_b = (const float*)d_in[8];
  const float* emb_glob_W = (const float*)d_in[9];
  const float* emb_glob_b = (const float*)d_in[10];
  const float* edge_W0 = (const float*)d_in[11];
  const float* edge_b0 = (const float*)d_in[12];
  const float* edge_W1 = (const float*)d_in[13];
  const float* edge_b1 = (const float*)d_in[14];
  const float* node_W0 = (const float*)d_in[15];
  const float* node_b0 = (const float*)d_in[16];
  const float* node_W1 = (const float*)d_in[17];
  const float* node_b1 = (const float*)d_in[18];
  const float* glob_W0 = (const float*)d_in[19];
  const float* glob_b0 = (const float*)d_in[20];
  const float* glob_W1 = (const float*)d_in[21];
  const float* glob_b1 = (const float*)d_in[22];
  const float* ln_node_s = (const float*)d_in[23];
  const float* ln_node_b = (const float*)d_in[24];
  const float* ln_edge_s = (const float*)d_in[25];
  const float* ln_edge_b = (const float*)d_in[26];
  const float* ln_glob_s = (const float*)d_in[27];
  const float* ln_glob_b = (const float*)d_in[28];
  const float* dec_W = (const float*)d_in[29];
  const float* dec_b = (const float*)d_in[30];
  (void)n_in; (void)out_size; (void)ws_size;

  const int N = in_sizes[0] / 64;   // 50000
  const int E = in_sizes[1] / 32;   // 500000

  // Live state lives directly in d_out: [nodes | edges | globals_out].
  float* nodes = (float*)d_out;
  float* edges = nodes + (size_t)N * LAT;
  float* gout  = edges + (size_t)E * LAT;

  // Workspace carve-up (256B aligned slabs).
  char* wsp = (char*)d_ws;
  auto carve = [&](size_t bytes) -> char* {
    char* p = wsp; wsp += (bytes + 255) & ~(size_t)255; return p;
  };
  float*  e_new    = (float*)carve((size_t)E * LAT * 4);
  float*  sent     = (float*)carve((size_t)N * LAT * 4);   // sent/recv/aggs
  float*  recv     = (float*)carve((size_t)N * LAT * 4);   // contiguous so one
  float*  node_agg = (float*)carve(LAT * 4);               // zero-fill covers
  float*  edge_agg = (float*)carve(LAT * 4);               // all four
  float*  n_new    = (float*)carve((size_t)N * LAT * 4);
  float*  glob     = (float*)carve(LAT * 4);
  bf16_t* embN_t   = (bf16_t*)carve((size_t)LAT * 64 * 2);
  bf16_t* embE_t   = (bf16_t*)carve((size_t)LAT * 32 * 2);
  bf16_t *eW0t[2], *eW1t[2], *nW0t[2], *nW1t[2];
  for (int s = 0; s < 2; ++s) {
    eW0t[s] = (bf16_t*)carve((size_t)LAT * 512 * 2);
    eW1t[s] = (bf16_t*)carve((size_t)LAT * LAT * 2);
    nW0t[s] = (bf16_t*)carve((size_t)LAT * 512 * 2);
    nW1t[s] = (bf16_t*)carve((size_t)LAT * LAT * 2);
  }

  const dim3 blk(256);
  auto cdiv = [](long a, long b) { return (unsigned)((a + b - 1) / b); };

  // ---- weight prep: cast+transpose to bf16 [N_out, K] --------------------
  gn_cast_transpose<<<cdiv(64 * LAT, 256), blk, 0, stream>>>(emb_node_W, embN_t, 64);
  gn_cast_transpose<<<cdiv(32 * LAT, 256), blk, 0, stream>>>(emb_edge_W, embE_t, 32);
  for (int s = 0; s < 2; ++s) {
    gn_cast_transpose<<<cdiv(512 * LAT, 256), blk, 0, stream>>>(edge_W0 + (size_t)s * 512 * LAT, eW0t[s], 512);
    gn_cast_transpose<<<cdiv(LAT * LAT, 256), blk, 0, stream>>>(edge_W1 + (size_t)s * LAT * LAT, eW1t[s], LAT);
    gn_cast_transpose<<<cdiv(512 * LAT, 256), blk, 0, stream>>>(node_W0 + (size_t)s * 512 * LAT, nW0t[s], 512);
    gn_cast_transpose<<<cdiv(LAT * LAT, 256), blk, 0, stream>>>(node_W1 + (size_t)s * LAT * LAT, nW1t[s], LAT);
  }

  // ---- embeddings --------------------------------------------------------
  gn_embed<<<cdiv(N, 16), blk, 0, stream>>>(node_feats, embN_t, emb_node_b, nodes, N, 64);
  gn_embed<<<cdiv(E, 16), blk, 0, stream>>>(edge_feats, embE_t, emb_edge_b, edges, E, 32);
  gn_glob_embed<<<1, 128, 0, stream>>>(globals_, emb_glob_W, emb_glob_b, glob);

  // ---- message-passing steps --------------------------------------------
  for (int s = 0; s < 2; ++s) {
    const int zn = N * LAT * 2 + 2 * LAT;   // sent + recv + node_agg + edge_agg
    gn_zero<<<cdiv(zn, 256), blk, 0, stream>>>(sent, zn);

    gn_edge_mlp<<<cdiv(E, 16), blk, 0, stream>>>(
        edges, nodes, glob, senders, receivers,
        eW0t[s], edge_b0 + s * LAT, eW1t[s], edge_b1 + s * LAT,
        e_new, sent, recv, edge_agg, E);

    gn_ln_residual<<<cdiv(E, 8), blk, 0, stream>>>(
        edges, e_new, ln_edge_s + s * LAT, ln_edge_b + s * LAT, E);

    gn_node_mlp<<<cdiv(N, 16), blk, 0, stream>>>(
        nodes, sent, recv, glob,
        nW0t[s], node_b0 + s * LAT, nW1t[s], node_b1 + s * LAT,
        n_new, node_agg, N);

    gn_glob_update<<<1, 128, 0, stream>>>(
        glob, node_agg, edge_agg,
        glob_W0 + (size_t)s * 384 * LAT, glob_b0 + s * LAT,
        glob_W1 + (size_t)s * LAT * LAT, glob_b1 + s * LAT,
        ln_glob_s + s * LAT, ln_glob_b + s * LAT);

    gn_ln_residual<<<cdiv(N, 8), blk, 0, stream>>>(
        nodes, n_new, ln_node_s + s * LAT, ln_node_b + s * LAT, N);
  }

  // ---- decode global output ---------------------------------------------
  gn_decode<<<1, 128, 0, stream>>>(glob, dec_W, dec_b, gout);
}
